// DimVariationalEmcoder_19894288515586
// MI455X (gfx1250) — matrix operand order repaired and checked by reading, not compiled, not run on hardware
//
#include <hip/hip_runtime.h>
#include <hip/hip_bf16.h>

// MI455X / gfx1250 variational graph encoder:
//   GCNConv(256->128) + LeakyReLU, then two TransformerConv(128->64) heads.
// All dense GEMMs use v_wmma_f32_16x16x32_bf16 with fp32 accumulate.
// A operands are pre-converted to bf16; B operands are pre-packed into WMMA
// fragment order so the inner loop is 4x global_load_b128 + 1 wmma per K=32.

typedef __attribute__((ext_vector_type(16))) __bf16 v16bf;
typedef __attribute__((ext_vector_type(8)))  __bf16 v8bf;
typedef __attribute__((ext_vector_type(8)))  float  v8f;

__device__ __forceinline__ __bf16 f2bf(float f) {
  unsigned int u = __float_as_uint(f);
  unsigned int r = (u + 0x7FFFu + ((u >> 16) & 1u)) >> 16; // round-to-nearest-even
  unsigned short s = (unsigned short)r;
  __bf16 b;
  __builtin_memcpy(&b, &s, 2);
  return b;
}

// ---------------------------------------------------------------------------
// WMMA GEMM: C[M x Nn] = A_bf16[M x K] * Bpack[K x Nn] + bias.
// A: row-major bf16. B: fragment-packed: [ntile][K/32][lane][16 bf16].
// One wave per 16x16 tile, 8 waves per block (16x128 strip).
// Requires M%16==0, K%32==0, Nn%16==0 (true here: 50000/16, 256|128, 128|256).
// ---------------------------------------------------------------------------
template <int K>
__global__ void wmma_gemm_bf16(const __bf16* __restrict__ A,
                               const __bf16* __restrict__ Bp,
                               const float* __restrict__ bias,
                               float* __restrict__ C,
                               int M, int Nn) {
  const int lane = threadIdx.x & 31;
  const int wave = threadIdx.x >> 5;
  const int m0 = blockIdx.x * 16;
  const int n0 = blockIdx.y * 128 + wave * 16;
  if (n0 >= Nn || m0 >= M) return; // wave-uniform: EXEC stays all-ones

  const int mA    = lane & 15;        // A frag: both lane halves map M=0..15
  const int kselA = (lane >> 4) * 16; // upper half holds K+16..31 of each 32-block
  const int nB    = lane & 15;

  const __bf16* __restrict__ Arow  = A + (size_t)(m0 + mA) * K + kselA;
  const __bf16* __restrict__ Bbase = Bp + ((size_t)(n0 >> 4) * (K / 32)) * 512 + lane * 16;

  v8f acc = {0.f, 0.f, 0.f, 0.f, 0.f, 0.f, 0.f, 0.f};
#pragma unroll
  for (int kb = 0; kb < K / 32; ++kb) {
    const v8bf alo = *(const v8bf*)(Arow + kb * 32);
    const v8bf ahi = *(const v8bf*)(Arow + kb * 32 + 8);
    const v8bf blo = *(const v8bf*)(Bbase + (size_t)kb * 512);
    const v8bf bhi = *(const v8bf*)(Bbase + (size_t)kb * 512 + 8);
    const v16bf a = __builtin_shufflevector(alo, ahi, 0, 1, 2, 3, 4, 5, 6, 7,
                                            8, 9, 10, 11, 12, 13, 14, 15);
    const v16bf b = __builtin_shufflevector(blo, bhi, 0, 1, 2, 3, 4, 5, 6, 7,
                                            8, 9, 10, 11, 12, 13, 14, 15);
    acc = __builtin_amdgcn_wmma_f32_16x16x32_bf16(false, a, false, b,
                                                  (short)0, acc, false, false);
  }

  const float badd = bias ? bias[n0 + nB] : 0.0f;
  const int mhalf = (lane >> 4) << 3; // lanes 16-31 write rows M=8..15
#pragma unroll
  for (int r = 0; r < 8; ++r)
    C[(size_t)(m0 + mhalf + r) * Nn + (n0 + nB)] = acc[r] + badd;
}

// ---------------------------------------------------------------------------
// Packing / conversion kernels
// ---------------------------------------------------------------------------
__global__ void k_cvt_bf16(const float* __restrict__ in, __bf16* __restrict__ out, long n) {
  long t = (long)blockIdx.x * blockDim.x + threadIdx.x;
  if (t < n) out[t] = f2bf(in[t]);
}

// Fragment-pack a single fp32 K x Nn matrix into [ntile][K/32][lane][16] bf16.
__global__ void k_frag_pack_f32(const float* __restrict__ W, __bf16* __restrict__ Bp,
                                int K, int Nn) {
  long t = (long)blockIdx.x * blockDim.x + threadIdx.x;
  if (t >= (long)K * Nn) return;
  const long c = t >> 9;          // 512-element chunk = one (ntile, kblock)
  const int r = (int)(t & 511);
  const int lane = r >> 4, jj = r & 15;
  const int kblocks = K / 32;
  const int nt = (int)(c / kblocks), kb = (int)(c % kblocks);
  const int k = kb * 32 + ((lane >> 4) << 4) + jj;
  const int n = (nt << 4) + (lane & 15);
  Bp[t] = f2bf(W[(size_t)k * Nn + n]);
}

// Fragment-pack fused [Wq|Wk|Wv|Ws] (each K x D) as a virtual K x 4D matrix.
__global__ void k_frag_pack_qkvs(const float* __restrict__ Wq, const float* __restrict__ Wk,
                                 const float* __restrict__ Wv, const float* __restrict__ Ws,
                                 __bf16* __restrict__ Bp, int K, int D) {
  const int Nn = 4 * D;
  long t = (long)blockIdx.x * blockDim.x + threadIdx.x;
  if (t >= (long)K * Nn) return;
  const long c = t >> 9;
  const int r = (int)(t & 511);
  const int lane = r >> 4, jj = r & 15;
  const int kblocks = K / 32;
  const int nt = (int)(c / kblocks), kb = (int)(c % kblocks);
  const int k = kb * 32 + ((lane >> 4) << 4) + jj;
  const int n = (nt << 4) + (lane & 15);
  const int blk = n / D, nn = n - blk * D;
  const float* Wsel = (blk == 0) ? Wq : (blk == 1) ? Wk : (blk == 2) ? Wv : Ws;
  Bp[t] = f2bf(Wsel[(size_t)k * D + nn]);
}

__global__ void k_pack_bias(const float* __restrict__ bq, const float* __restrict__ bk,
                            const float* __restrict__ bv, const float* __restrict__ bs,
                            float* __restrict__ bcat, int D) {
  int t = blockIdx.x * blockDim.x + threadIdx.x;
  if (t >= 4 * D) return;
  const int blk = t / D, nn = t - blk * D;
  const float* bsel = (blk == 0) ? bq : (blk == 1) ? bk : (blk == 2) ? bv : bs;
  bcat[t] = bsel[nn];
}

// ---------------------------------------------------------------------------
// Elementwise / graph kernels
// ---------------------------------------------------------------------------
__global__ void k_fill_f32(float* __restrict__ p, float v, long n) {
  long t = (long)blockIdx.x * blockDim.x + threadIdx.x;
  if (t < n) p[t] = v;
}

__global__ void k_fill_u32(unsigned int* __restrict__ p, unsigned int v, long n) {
  long t = (long)blockIdx.x * blockDim.x + threadIdx.x;
  if (t < n) p[t] = v;
}

__global__ void k_deg_edges(const int* __restrict__ dst, float* __restrict__ deg, long E) {
  long e = (long)blockIdx.x * blockDim.x + threadIdx.x;
  if (e < E) atomicAdd(&deg[dst[e]], 1.0f);
}

__global__ void k_rsqrt_inplace(float* __restrict__ p, long n) {
  long t = (long)blockIdx.x * blockDim.x + threadIdx.x;
  if (t < n) p[t] = rsqrtf(p[t]); // deg >= 1 always (self loop)
}

__global__ void k_edge_norm(const int* __restrict__ src, const int* __restrict__ dst,
                            const float* __restrict__ dinv, float* __restrict__ norm, long E) {
  long e = (long)blockIdx.x * blockDim.x + threadIdx.x;
  if (e < E) norm[e] = dinv[src[e]] * dinv[dst[e]];
}

__global__ void k_gcn_self(const float* __restrict__ h0, const float* __restrict__ dinv,
                           float* __restrict__ hagg, long N, int H) {
  long t = (long)blockIdx.x * blockDim.x + threadIdx.x;
  if (t >= N * (long)H) return;
  const long i = t / H;
  const float dv = dinv[i];
  hagg[t] = h0[t] * dv * dv; // self-loop contribution
}

__global__ void k_gcn_scatter(const int* __restrict__ src, const int* __restrict__ dst,
                              const float* __restrict__ h0, const float* __restrict__ norm,
                              float* __restrict__ hagg, long E, int H) {
  long t = (long)blockIdx.x * blockDim.x + threadIdx.x;
  if (t >= E * (long)H) return;
  const long e = t / H;
  const int j = (int)(t - e * H);
  atomicAdd(&hagg[(size_t)dst[e] * H + j], h0[(size_t)src[e] * H + j] * norm[e]);
}

// h = LeakyReLU(hagg + b), written directly as bf16 (next GEMM's A operand).
__global__ void k_bias_leaky_bf(const float* __restrict__ hagg, const float* __restrict__ b,
                                __bf16* __restrict__ h, long N, int H) {
  long t = (long)blockIdx.x * blockDim.x + threadIdx.x;
  if (t >= N * (long)H) return;
  const int j = (int)(t % H);
  const float v = hagg[t] + b[j];
  h[t] = f2bf((v >= 0.0f) ? v : 0.01f * v);
}

// alpha[e] = dot(q[dst], k[src]) / sqrt(D); QKVS row = [q | k | v | s], stride 4D
__global__ void k_alpha(const int* __restrict__ src, const int* __restrict__ dst,
                        const float* __restrict__ qkvs, float* __restrict__ alpha,
                        long E, int D, float scale) {
  long e = (long)blockIdx.x * blockDim.x + threadIdx.x;
  if (e >= E) return;
  const float4* q4 = (const float4*)(qkvs + (size_t)dst[e] * 4 * D);
  const float4* k4 = (const float4*)(qkvs + (size_t)src[e] * 4 * D + D);
  float s = 0.0f;
  for (int j = 0; j < D / 4; ++j) {
    const float4 a = q4[j], b = k4[j];
    s += a.x * b.x + a.y * b.y + a.z * b.z + a.w * b.w;
  }
  alpha[e] = s * scale;
}

// Order-preserving float->uint encoding so atomicMax(u32) implements float max.
__device__ __forceinline__ unsigned int enc_f32(float f) {
  unsigned int u = __float_as_uint(f);
  return (u & 0x80000000u) ? ~u : (u | 0x80000000u);
}
__device__ __forceinline__ float dec_f32(unsigned int e) {
  e = (e & 0x80000000u) ? (e & 0x7FFFFFFFu) : ~e;
  return __uint_as_float(e);
}

__global__ void k_amax(const int* __restrict__ dst, const float* __restrict__ alpha,
                       unsigned int* __restrict__ amax, long E) {
  long e = (long)blockIdx.x * blockDim.x + threadIdx.x;
  if (e < E) atomicMax(&amax[dst[e]], enc_f32(alpha[e]));
}

__global__ void k_expsum(const int* __restrict__ dst, float* __restrict__ alpha,
                         const unsigned int* __restrict__ amax, float* __restrict__ denom,
                         long E) {
  long e = (long)blockIdx.x * blockDim.x + threadIdx.x;
  if (e >= E) return;
  const int d = dst[e];
  const float ex = __expf(alpha[e] - dec_f32(amax[d]));
  alpha[e] = ex;
  atomicAdd(&denom[d], ex);
}

__global__ void k_out_init(const float* __restrict__ qkvs, float* __restrict__ out,
                           long N, int D) {
  long t = (long)blockIdx.x * blockDim.x + threadIdx.x;
  if (t >= N * (long)D) return;
  const long i = t / D;
  const int d = (int)(t - i * D);
  out[t] = qkvs[(size_t)i * 4 * D + 3 * D + d]; // skip projection s (bias included)
}

__global__ void k_attn_scatter(const int* __restrict__ src, const int* __restrict__ dst,
                               const float* __restrict__ qkvs, const float* __restrict__ alpha,
                               const float* __restrict__ denom, float* __restrict__ out,
                               long E, int D) {
  long t = (long)blockIdx.x * blockDim.x + threadIdx.x;
  if (t >= E * (long)D) return;
  const long e = t / D;
  const int d = (int)(t - e * D);
  const int di = dst[e];
  const float w = alpha[e] / (denom[di] + 1e-16f);
  atomicAdd(&out[(size_t)di * D + d], qkvs[(size_t)src[e] * 4 * D + 2 * D + d] * w);
}

__global__ void k_clamp(float* __restrict__ p, float mx, long n) {
  long t = (long)blockIdx.x * blockDim.x + threadIdx.x;
  if (t < n) p[t] = fminf(p[t], mx);
}

// ---------------------------------------------------------------------------
extern "C" void kernel_launch(void* const* d_in, const int* in_sizes, int n_in,
                              void* d_out, int out_size, void* d_ws, size_t ws_size,
                              hipStream_t stream) {
  const float* x     = (const float*)d_in[0];
  const int*   ei    = (const int*)d_in[1]; // [2, E] int32: row0=src, row1=dst
  const float* W_gcn = (const float*)d_in[2];
  const float* b_gcn = (const float*)d_in[3];
  const float* Wq_mu = (const float*)d_in[4];  const float* bq_mu = (const float*)d_in[5];
  const float* Wk_mu = (const float*)d_in[6];  const float* bk_mu = (const float*)d_in[7];
  const float* Wv_mu = (const float*)d_in[8];  const float* bv_mu = (const float*)d_in[9];
  const float* Ws_mu = (const float*)d_in[10]; const float* bs_mu = (const float*)d_in[11];
  const float* Wq_ls = (const float*)d_in[12]; const float* bq_ls = (const float*)d_in[13];
  const float* Wk_ls = (const float*)d_in[14]; const float* bk_ls = (const float*)d_in[15];
  const float* Wv_ls = (const float*)d_in[16]; const float* bv_ls = (const float*)d_in[17];
  const float* Ws_ls = (const float*)d_in[18]; const float* bs_ls = (const float*)d_in[19];

  const int  H = in_sizes[3];                // 128
  const int  D = in_sizes[5];                // 64
  const int  F = in_sizes[2] / H;            // 256
  const long N = (long)in_sizes[0] / F;      // 50000 (divisible by 16)
  const long E = (long)in_sizes[1] / 2;      // 800000
  const int* src = ei;
  const int* dst = ei + E;

  // Carve workspace (256B aligned regions).
  char* w = (char*)d_ws;
  auto carve = [&](size_t bytes) -> void* {
    void* p = (void*)w;
    w += (bytes + 255) & ~(size_t)255;
    return p;
  };
  float*        deg   = (float*)carve((size_t)N * 4);            // degree -> dinv
  unsigned int* amax  = (unsigned int*)carve((size_t)N * 4);     // encoded segment max
  float*        denom = (float*)carve((size_t)N * 4);
  float*        h0    = (float*)carve((size_t)N * H * 4);        // xW (fp32)
  float*        hagg  = (float*)carve((size_t)N * H * 4);
  float*        qkvs  = (float*)carve((size_t)N * 4 * D * 4);    // [q|k|v|s] per node
  float*        ework = (float*)carve((size_t)E * 4);            // edge norm / alpha / ex
  __bf16*       x_bf  = (__bf16*)carve((size_t)N * F * 2);       // bf16 copy of x
  __bf16*       h_bf  = (__bf16*)carve((size_t)N * H * 2);       // bf16 hidden
  __bf16*       Wg_p  = (__bf16*)carve((size_t)F * H * 2);       // frag-packed weights
  __bf16*       Wc_mu = (__bf16*)carve((size_t)H * 4 * D * 2);
  __bf16*       Wc_ls = (__bf16*)carve((size_t)H * 4 * D * 2);
  float*        bc_mu = (float*)carve((size_t)4 * D * 4);
  float*        bc_ls = (float*)carve((size_t)4 * D * 4);
  (void)ws_size; (void)n_in; (void)out_size;

  const int TB = 256;
  auto nb = [](long n) { return (unsigned)((n + 255) / 256); };

  auto launch_gemm = [&](const __bf16* A, const __bf16* Bp, const float* bias,
                         float* C, long M, int K, int Nn) {
    dim3 g((unsigned)((M + 15) / 16), (unsigned)((Nn + 127) / 128));
    if (K == 256)
      wmma_gemm_bf16<256><<<g, 256, 0, stream>>>(A, Bp, bias, C, (int)M, Nn);
    else if (K == 128)
      wmma_gemm_bf16<128><<<g, 256, 0, stream>>>(A, Bp, bias, C, (int)M, Nn);
    else if (K == 64)
      wmma_gemm_bf16<64><<<g, 256, 0, stream>>>(A, Bp, bias, C, (int)M, Nn);
    else
      wmma_gemm_bf16<32><<<g, 256, 0, stream>>>(A, Bp, bias, C, (int)M, Nn);
  };

  // --- operand prep: bf16 conversion + WMMA fragment packing ---
  k_cvt_bf16<<<nb(N * F), TB, 0, stream>>>(x, x_bf, N * F);
  k_frag_pack_f32<<<nb((long)F * H), TB, 0, stream>>>(W_gcn, Wg_p, F, H);
  k_frag_pack_qkvs<<<nb((long)H * 4 * D), TB, 0, stream>>>(Wq_mu, Wk_mu, Wv_mu, Ws_mu,
                                                           Wc_mu, H, D);
  k_frag_pack_qkvs<<<nb((long)H * 4 * D), TB, 0, stream>>>(Wq_ls, Wk_ls, Wv_ls, Ws_ls,
                                                           Wc_ls, H, D);
  k_pack_bias<<<nb(4 * D), TB, 0, stream>>>(bq_mu, bk_mu, bv_mu, bs_mu, bc_mu, D);
  k_pack_bias<<<nb(4 * D), TB, 0, stream>>>(bq_ls, bk_ls, bv_ls, bs_ls, bc_ls, D);

  // --- GCN degree normalization ---
  k_fill_f32<<<nb(N), TB, 0, stream>>>(deg, 1.0f, N);             // self loop
  k_deg_edges<<<nb(E), TB, 0, stream>>>(dst, deg, E);
  k_rsqrt_inplace<<<nb(N), TB, 0, stream>>>(deg, N);              // deg := d^-1/2

  // --- h0 = x @ W_gcn (WMMA bf16, K=256 fully unrolled -> 8 wmma) ---
  launch_gemm(x_bf, Wg_p, nullptr, h0, N, F, H);

  // --- GCN aggregation + bias + LeakyReLU (-> bf16 h) ---
  k_edge_norm<<<nb(E), TB, 0, stream>>>(src, dst, deg, ework, E);
  k_gcn_self<<<nb(N * H), TB, 0, stream>>>(h0, deg, hagg, N, H);
  k_gcn_scatter<<<nb(E * H), TB, 0, stream>>>(src, dst, h0, ework, hagg, E, H);
  k_bias_leaky_bf<<<nb(N * H), TB, 0, stream>>>(hagg, b_gcn, h_bf, N, H);

  // --- two TransformerConv heads: mu (layer 0), logstd (layer 1) ---
  float* outp = (float*)d_out;
  const float scale = 1.0f / sqrtf((float)D);
  for (int layer = 0; layer < 2; ++layer) {
    const __bf16* Wc = layer ? Wc_ls : Wc_mu;
    const float*  bc = layer ? bc_ls : bc_mu;
    float* out = outp + (size_t)layer * N * D;

    launch_gemm(h_bf, Wc, bc, qkvs, N, H, 4 * D); // K=128 -> 4 wmma unrolled

    k_alpha<<<nb(E), TB, 0, stream>>>(src, dst, qkvs, ework, E, D, scale);
    k_fill_u32<<<nb(N), TB, 0, stream>>>(amax, 0x007FFFFFu, N);   // enc(-inf)
    k_amax<<<nb(E), TB, 0, stream>>>(dst, ework, amax, E);
    k_fill_f32<<<nb(N), TB, 0, stream>>>(denom, 0.0f, N);
    k_expsum<<<nb(E), TB, 0, stream>>>(dst, ework, amax, denom, E);
    k_out_init<<<nb(N * D), TB, 0, stream>>>(qkvs, out, N, D);
    k_attn_scatter<<<nb(E * D), TB, 0, stream>>>(src, dst, qkvs, ework, denom, out, E, D);
  }

  // logstd = min(logstd, 10)
  k_clamp<<<nb(N * D), TB, 0, stream>>>(outp + (size_t)N * D, 10.0f, N * D);
}
